// Attention_5978594476296
// MI455X (gfx1250) — matrix operand order repaired and checked by reading, not compiled, not run on hardware
//
#include <hip/hip_runtime.h>
#include <hip/hip_bf16.h>

// ---------------------------------------------------------------------------
// MHA forward for gfx1250 (MI455X): f16 WMMA GEMMs + flash attention,
// with Tensor-Data-Mover (TDM) tile staging into LDS.
// B=2, S=2048, D=1024, H=16, DK=64.
// ---------------------------------------------------------------------------

typedef __attribute__((ext_vector_type(16))) _Float16     v16h;
typedef __attribute__((ext_vector_type(4)))  _Float16     v4h;
typedef __attribute__((ext_vector_type(8)))  float        v8f;
typedef __attribute__((ext_vector_type(4)))  unsigned int u32x4;
typedef __attribute__((ext_vector_type(8)))  int          i32x8;
typedef __attribute__((ext_vector_type(4)))  int          i32x4;

#define BB   2
#define SS   2048
#define DD   1024
#define HH   16
#define DKK  64

#if __has_builtin(__builtin_amdgcn_tensor_load_to_lds) && \
    __has_builtin(__builtin_amdgcn_s_wait_tensorcnt)
#define USE_TDM 1
#else
#define USE_TDM 0
#endif

union Frag16 { v16h v; uint4 u4[2]; };

// A-fragment (16x32 f16, MxK) at k-offset kk within an LDS tile row:
// lane L<16 -> row M=L, K = kk + (0..7 & 16..23);
// lane L>=16 -> row M=L-16, K = kk + (8..15 & 24..31).   (ISA 7.12.2)
__device__ __forceinline__ v16h load_frag_a(const _Float16* p, int stride, int lane, int kk) {
  Frag16 f;
  const int r  = lane & 15;
  const int hf = (lane >> 4) & 1;
  const _Float16* row = p + (size_t)r * stride + kk + hf * 8;
  f.u4[0] = *reinterpret_cast<const uint4*>(row);
  f.u4[1] = *reinterpret_cast<const uint4*>(row + 16);
  return f.v;
}

// B-fragment (32x16 f16, KxN) from Bt (=B transposed, NxK row-major):
// lane L<16 -> col N=L, K = kk + (0..7 & 16..23); lane L>=16 -> col N=L-16, K = kk + (8..15 & 24..31).
__device__ __forceinline__ v16h load_frag_b(const _Float16* p, int stride, int lane, int kk) {
  Frag16 f;
  const int r  = lane & 15;
  const int hf = (lane >> 4) & 1;
  const _Float16* row = p + (size_t)r * stride + kk + hf * 8;
  f.u4[0] = *reinterpret_cast<const uint4*>(row);
  f.u4[1] = *reinterpret_cast<const uint4*>(row + 16);
  return f.v;
}

__device__ __forceinline__ v8f wmma16x16x32(v16h a, v16h b, v8f c) {
  return __builtin_amdgcn_wmma_f32_16x16x32_f16(false, a, false, b, (short)0, c,
                                                false, false);
}

#if USE_TDM
// ---------------------------------------------------------------------------
// TDM 2-D tile load: Global -> LDS.  Builds a D# descriptor per
// CDNA5 ISA (data_size = 2 bytes, 2-D tile with
// dim0 stride).  Generic LDS pointers on AMDGPU carry the wave-relative LDS
// byte offset in their low 32 bits (flat-aperture model), which is what the
// descriptor's lds_addr wants.  This toolchain's builtin takes the 6-arg
// form: (g0 v4u, g1 v8i, g2 v4i, g3 v4i, v8i, cpol).
// ---------------------------------------------------------------------------
__device__ __forceinline__ void tdm_load_2d_f16(const _Float16* lds_dst,
                                                const _Float16* gsrc,
                                                unsigned tile_d0, unsigned tile_d1,
                                                unsigned long long stride0,
                                                unsigned long long tdim0,
                                                unsigned long long tdim1) {
  const unsigned long long ga = (unsigned long long)(uintptr_t)gsrc;
  const unsigned lds_off = (unsigned)(uintptr_t)lds_dst;
  u32x4 g0;
  g0[0] = 1u;                                              // count=1, user desc
  g0[1] = lds_off;                                         // lds_addr (bytes)
  g0[2] = (unsigned)(ga & 0xffffffffu);                    // global_addr[31:0]
  g0[3] = (unsigned)((ga >> 32) & 0x01ffffffu) | 0x80000000u; // addr[56:32] | type=2
  i32x8 g1;
  g1[0] = (int)(1u << 16);                                 // wg_mask=0, data_size=2B
  g1[1] = (int)((tdim0 & 0xffffu) << 16);                  // abar=0 | tensor_dim0[15:0]
  g1[2] = (int)(((tdim0 >> 16) & 0xffffu) | ((tdim1 & 0xffffu) << 16));
  g1[3] = (int)(((tdim1 >> 16) & 0xffffu) | ((tile_d0 & 0xffffu) << 16));
  g1[4] = (int)(tile_d1 & 0xffffu);                        // tile_dim1 | tile_dim2=0
  g1[5] = (int)(stride0 & 0xffffffffu);                    // dim0 stride [31:0]
  g1[6] = (int)((stride0 >> 32) & 0xffffu);                // dim0 stride [47:32]
  g1[7] = 0;
  const i32x4 z4 = {0, 0, 0, 0};
  const i32x8 z8 = {0, 0, 0, 0, 0, 0, 0, 0};
  __builtin_amdgcn_tensor_load_to_lds(g0, g1, z4, z4, z8, 0);
}
#endif

// ---------------------------------------------------------------------------
// f32 -> f16 conversion, 4-wide.
// ---------------------------------------------------------------------------
__global__ void __launch_bounds__(256)
cvt_f32_f16_x4(const float* __restrict__ src, _Float16* __restrict__ dst, int n4) {
  int i = blockIdx.x * blockDim.x + threadIdx.x;
  if (i < n4) {
    float4 v = reinterpret_cast<const float4*>(src)[i];
    v4h h = { (_Float16)v.x, (_Float16)v.y, (_Float16)v.z, (_Float16)v.w };
    reinterpret_cast<v4h*>(dst)[i] = h;
  }
}

// ---------------------------------------------------------------------------
// Projection GEMM: Y[m][n] = sum_k X[m][k] * W[n][k] + bias[n]
// M=4096, N=1024, K=1024.  Tile 64x64, K-step 64, 4 waves (128 thr):
// 8 WMMAs per wave per staged tile.
// MODE 0: write f16 into head layout  [b][h][s][dk]
// MODE 1: write f16 into transposed   [b][h][dk][s]   (for V)
// MODE 2: write f32 into plain [m][n] (final output)
// ---------------------------------------------------------------------------
template <int MODE>
__global__ void __launch_bounds__(128)
proj_gemm(const _Float16* __restrict__ X, const _Float16* __restrict__ W,
          const float* __restrict__ bias, void* __restrict__ outp) {
  __shared__ alignas(16) _Float16 sA[64 * 64];
  __shared__ alignas(16) _Float16 sB[64 * 64];

  const int tid  = threadIdx.x;
  const int lane = tid & 31;
  const int w    = tid >> 5;
  const int mt   = blockIdx.x * 64;
  const int nt   = blockIdx.y * 64;

  v8f vzero = {};
  v8f acc[4];
#pragma unroll
  for (int j = 0; j < 4; ++j) acc[j] = vzero;

  for (int k0 = 0; k0 < DD; k0 += 64) {
    __syncthreads();
#if USE_TDM
    if (w == 0) {
      // One TDM descriptor per 64x64 f16 tile (stride DD elements).
      tdm_load_2d_f16(sA, X + (size_t)mt * DD + k0, 64, 64, DD, DD, 4096);
      tdm_load_2d_f16(sB, W + (size_t)nt * DD + k0, 64, 64, DD, DD, 1024);
      __builtin_amdgcn_s_wait_tensorcnt(0);
    }
#else
#pragma unroll
    for (int c = tid; c < 512; c += 128) {   // 512 16-byte chunks per tile
      const int r = c >> 3, off = (c & 7) * 8;
      *reinterpret_cast<uint4*>(sA + r * 64 + off) =
          *reinterpret_cast<const uint4*>(X + (size_t)(mt + r) * DD + k0 + off);
      *reinterpret_cast<uint4*>(sB + r * 64 + off) =
          *reinterpret_cast<const uint4*>(W + (size_t)(nt + r) * DD + k0 + off);
    }
#endif
    __syncthreads();

#pragma unroll
    for (int kk = 0; kk < 64; kk += 32) {
      const v16h a = load_frag_a(sA + (w * 16) * 64, 64, lane, kk);
#pragma unroll
      for (int j = 0; j < 4; ++j) {
        const v16h b = load_frag_b(sB + (j * 16) * 64, 64, lane, kk);
        acc[j] = wmma16x16x32(a, b, acc[j]);
      }
    }
  }

  // Epilogue. C/D layout: element [m][n] -> lane (m/8)*16 + n, vgpr m%8.
  const int ln = lane & 15;
  const int mb = (lane >> 4) * 8;
#pragma unroll
  for (int j = 0; j < 4; ++j) {
#pragma unroll
    for (int i = 0; i < 8; ++i) {
      const int gm = mt + w * 16 + mb + i;  // global row  (b*S + s)
      const int gn = nt + j * 16 + ln;      // global col  (h*DK + dk)
      const float val = acc[j][i] + bias[gn];
      if (MODE == 2) {
        reinterpret_cast<float*>(outp)[(size_t)gm * DD + gn] = val;
      } else {
        const int bb = gm >> 11, s = gm & (SS - 1);
        const int h = gn >> 6, dk = gn & (DKK - 1);
        _Float16* o = reinterpret_cast<_Float16*>(outp);
        if (MODE == 0)
          o[(((size_t)bb * HH + h) * SS + s) * DKK + dk] = (_Float16)val;
        else
          o[(((size_t)bb * HH + h) * DKK + dk) * SS + s] = (_Float16)val;
      }
    }
  }
}

// ---------------------------------------------------------------------------
// Flash attention: per WG = 64 query rows of one (b,h); 4 waves x 16 rows.
// Key tiles of 32; online softmax; 8 WMMAs / wave / key tile.
// ---------------------------------------------------------------------------
__global__ void __launch_bounds__(128)
flash_attn(const _Float16* __restrict__ q16, const _Float16* __restrict__ k16,
           const _Float16* __restrict__ vt16, const int* __restrict__ mask,
           _Float16* __restrict__ o16) {
  __shared__ alignas(16) _Float16 sK[32 * 64];        // 32 keys x 64 dk
  __shared__ alignas(16) _Float16 sV[64 * 32];        // 64 dk x 32 keys (V^T)
  __shared__ alignas(16) _Float16 sP[4][16 * 32];     // per-wave P staging

  const int tid  = threadIdx.x;
  const int lane = tid & 31;
  const int w    = tid >> 5;
  const int h    = blockIdx.y;
  const int bz   = blockIdx.z;
  const int qbase = blockIdx.x * 64 + w * 16;
  const size_t bh = (size_t)bz * HH + h;

  const int ln = lane & 15;       // A-frag row / C-frag col / B-frag col
  const int hf = lane >> 4;

  // Q A-fragments, pinned in VGPRs for the whole kernel (DK=64 -> 2 frags).
  const _Float16* qrow = q16 + (bh * SS + qbase + ln) * DKK;
  Frag16 aQ0, aQ1;
  aQ0.u4[0] = *reinterpret_cast<const uint4*>(qrow + hf * 8);
  aQ0.u4[1] = *reinterpret_cast<const uint4*>(qrow + 16 + hf * 8);
  aQ1.u4[0] = *reinterpret_cast<const uint4*>(qrow + 32 + hf * 8);
  aQ1.u4[1] = *reinterpret_cast<const uint4*>(qrow + 48 + hf * 8);

  float mrow[8], lrow[8];
  v8f vzero = {};
  v8f accO[4];
#pragma unroll
  for (int j = 0; j < 4; ++j) accO[j] = vzero;
#pragma unroll
  for (int i = 0; i < 8; ++i) { mrow[i] = -3.0e38f; lrow[i] = 0.0f; }

  const _Float16* Kbase = k16 + bh * SS * DKK;
  const _Float16* Vbase = vt16 + bh * DKK * SS;
  const float sc_qk = 0.125f;  // 1/sqrt(DK)

  for (int kt = 0; kt < SS; kt += 32) {
    __syncthreads();
#if USE_TDM
    if (w == 0) {
      // K tile: 32 rows x 64 dk (row stride DK); V^T tile: 64 rows x 32 keys
      // (row stride S).
      tdm_load_2d_f16(sK, Kbase + (size_t)kt * DKK, 64, 32, DKK, DKK, SS);
      tdm_load_2d_f16(sV, Vbase + kt, 32, 64, SS, SS, DKK);
      __builtin_amdgcn_s_wait_tensorcnt(0);
    }
#else
#pragma unroll
    for (int c = tid; c < 256; c += 128) {
      const int r = c >> 3, off = (c & 7) * 8;
      *reinterpret_cast<uint4*>(sK + r * 64 + off) =
          *reinterpret_cast<const uint4*>(Kbase + (size_t)(kt + r) * DKK + off);
    }
#pragma unroll
    for (int c = tid; c < 256; c += 128) {
      const int r = c >> 2, off = (c & 3) * 8;
      *reinterpret_cast<uint4*>(sV + r * 32 + off) =
          *reinterpret_cast<const uint4*>(Vbase + (size_t)r * SS + kt + off);
    }
#endif
    if (kt + 32 < SS) {   // speculative prefetch of next tile
      __builtin_prefetch(Kbase + (size_t)(kt + 32) * DKK + tid * 16, 0, 0);
      __builtin_prefetch(Vbase + (size_t)(tid >> 1) * SS + kt + 32, 0, 0);
    }
    __syncthreads();

    // S = Q * K^T  for 16 queries x 32 keys (2 N-tiles x 2 K-steps).
    v8f s0 = vzero, s1 = vzero;
    {
      v16h b0 = load_frag_b(sK, 64, lane, 0);
      s0 = wmma16x16x32(aQ0.v, b0, s0);
      v16h b1 = load_frag_b(sK, 64, lane, 32);
      s0 = wmma16x16x32(aQ1.v, b1, s0);
      v16h b2 = load_frag_b(sK + 16 * 64, 64, lane, 0);
      s1 = wmma16x16x32(aQ0.v, b2, s1);
      v16h b3 = load_frag_b(sK + 16 * 64, 64, lane, 32);
      s1 = wmma16x16x32(aQ1.v, b3, s1);
    }

    // Online softmax on C-layout: row m = hf*8 + i spans the 16 lanes of a half.
    float p0[8], p1[8];
#pragma unroll
    for (int i = 0; i < 8; ++i) {
      float x0 = s0[i] * sc_qk;
      float x1 = s1[i] * sc_qk;
      const int* mp = mask + ((size_t)bz * SS + (qbase + hf * 8 + i)) * SS + kt;
      if (mp[ln] == 0)      x0 = -1.0e9f;
      if (mp[16 + ln] == 0) x1 = -1.0e9f;
      float t = fmaxf(x0, x1);
#pragma unroll
      for (int d = 1; d < 16; d <<= 1) t = fmaxf(t, __shfl_xor(t, d, 32));
      const float nm = fmaxf(mrow[i], t);
      const float rescale = __expf(mrow[i] - nm);
      p0[i] = __expf(x0 - nm);
      p1[i] = __expf(x1 - nm);
      float rs = p0[i] + p1[i];
#pragma unroll
      for (int d = 1; d < 16; d <<= 1) rs += __shfl_xor(rs, d, 32);
      lrow[i] = lrow[i] * rescale + rs;
      mrow[i] = nm;
#pragma unroll
      for (int j = 0; j < 4; ++j) accO[j][i] *= rescale;
    }

    // P (f32, C-layout) -> f16 A-fragment via per-wave LDS staging.
    _Float16* pw = &sP[w][0];
#pragma unroll
    for (int i = 0; i < 8; ++i) {
      const int r = hf * 8 + i;
      pw[r * 32 + ln]      = (_Float16)p0[i];
      pw[r * 32 + 16 + ln] = (_Float16)p1[i];
    }
    // Same-wave LDS RAW: CDNA5 split counter wait (DScnt) before reload.
    asm volatile("s_wait_dscnt 0" ::: "memory");

    const v16h aP = load_frag_a(pw, 32, lane, 0);
#pragma unroll
    for (int j = 0; j < 4; ++j) {
      const v16h bvf = load_frag_b(sV + (j * 16) * 32, 32, lane, 0);
      accO[j] = wmma16x16x32(aP, bvf, accO[j]);
    }
  }

  // Finalize: O /= l, write f16 into [b][s][h*DK+dk] for the output GEMM.
#pragma unroll
  for (int i = 0; i < 8; ++i) {
    const float inv = 1.0f / lrow[i];
    const int s = qbase + hf * 8 + i;
#pragma unroll
    for (int j = 0; j < 4; ++j) {
      const int gn = h * DKK + j * 16 + ln;
      o16[((size_t)bz * SS + s) * DD + gn] = (_Float16)(accO[j][i] * inv);
    }
  }
}

// ---------------------------------------------------------------------------
// Host launcher.
// ---------------------------------------------------------------------------
extern "C" void kernel_launch(void* const* d_in, const int* in_sizes, int n_in,
                              void* d_out, int out_size, void* d_ws, size_t ws_size,
                              hipStream_t stream) {
  const float* query = (const float*)d_in[0];
  const int*   mask  = (const int*)d_in[1];
  const float* Wq    = (const float*)d_in[2];
  const float* bq    = (const float*)d_in[3];
  const float* Wk    = (const float*)d_in[4];
  const float* bk    = (const float*)d_in[5];
  const float* Wv    = (const float*)d_in[6];
  const float* bv    = (const float*)d_in[7];
  const float* Wo    = (const float*)d_in[8];
  const float* bo    = (const float*)d_in[9];

  char* ws = (char*)d_ws;
  const size_t MB = 1u << 20;
  _Float16* query16 = (_Float16*)(ws + 0 * MB);   // 8 MB : [4096][1024]
  _Float16* Wq16    = (_Float16*)(ws + 8 * MB);   // 2 MB
  _Float16* Wk16    = (_Float16*)(ws + 10 * MB);  // 2 MB
  _Float16* Wv16    = (_Float16*)(ws + 12 * MB);  // 2 MB
  _Float16* Wo16    = (_Float16*)(ws + 14 * MB);  // 2 MB
  _Float16* q16     = (_Float16*)(ws + 16 * MB);  // 8 MB : [b][h][s][dk]
  _Float16* k16     = (_Float16*)(ws + 24 * MB);  // 8 MB : [b][h][s][dk]
  _Float16* vt16    = (_Float16*)(ws + 32 * MB);  // 8 MB : [b][h][dk][s]
  _Float16* o16     = (_Float16*)(ws + 40 * MB);  // 8 MB : [4096][1024]

  const int nq4 = (BB * SS * DD) / 4;
  const int nw4 = (DD * DD) / 4;
  cvt_f32_f16_x4<<<(nq4 + 255) / 256, 256, 0, stream>>>(query, query16, nq4);
  cvt_f32_f16_x4<<<(nw4 + 255) / 256, 256, 0, stream>>>(Wq, Wq16, nw4);
  cvt_f32_f16_x4<<<(nw4 + 255) / 256, 256, 0, stream>>>(Wk, Wk16, nw4);
  cvt_f32_f16_x4<<<(nw4 + 255) / 256, 256, 0, stream>>>(Wv, Wv16, nw4);
  cvt_f32_f16_x4<<<(nw4 + 255) / 256, 256, 0, stream>>>(Wo, Wo16, nw4);

  dim3 pg((BB * SS) / 64, DD / 64);  // (64, 16)
  proj_gemm<0><<<pg, 128, 0, stream>>>(query16, Wq16, bq, (void*)q16);
  proj_gemm<0><<<pg, 128, 0, stream>>>(query16, Wk16, bk, (void*)k16);
  proj_gemm<1><<<pg, 128, 0, stream>>>(query16, Wv16, bv, (void*)vt16);

  flash_attn<<<dim3(SS / 64, HH, BB), 128, 0, stream>>>(q16, k16, vt16, mask, o16);

  proj_gemm<2><<<pg, 128, 0, stream>>>(o16, Wo16, bo, d_out);
}
